// DYSOpt_OTF_37177236914242
// MI455X (gfx1250) — matrix operand order repaired and checked by reading, not compiled, not run on hardware
//
#include <hip/hip_runtime.h>
#include <stdint.h>

// ---------------------------------------------------------------- types ----
typedef __bf16 bf16_t;
typedef __attribute__((ext_vector_type(16))) __bf16 v16bf;
typedef __attribute__((ext_vector_type(8)))  __bf16 v8bf;
typedef __attribute__((ext_vector_type(4)))  __bf16 v4bf;
typedef __attribute__((ext_vector_type(8)))  float  v8f;
typedef __attribute__((ext_vector_type(4)))  float  v4f;

// ------------------------------------------------------------- problem ----
#define BATCH    256
#define MM       128
#define NN       512
#define DDIM     640     // MM + NN
#define NITER    50
#define ALPHA_C  0.05f
#define TAU_C    1.0f
#define DECAY_C  10.0f

// padded LDS row strides (bf16 elements) -> rows rotate across 64 banks
#define LDA   (NN + 8)   // 520  (1040 B/row, 16B aligned)
#define LDAI  (MM + 8)   // 136  (272 B/row, 16B aligned)

// LDS layout, byte offsets (dynamic shared starts at LDS address 0)
#define OFF_A      0
#define OFF_AI     (OFF_A  + MM   * LDA  * 2)   // 133120
#define OFF_TBF    (OFF_AI + DDIM * LDAI * 2)   // 307200
#define OFF_RBF    (OFF_TBF + NN * 2)           // +1024
#define OFF_SOL    (OFF_RBF + MM * 2)           // +256
#define OFF_TTAIL  (OFF_SOL + DDIM * 4)         // +2560
#define OFF_C      (OFF_TTAIL + MM * 4)         // +512
#define OFF_BV     (OFF_C + DDIM * 4)           // +2560
#define LDS_BYTES  (OFF_BV + MM * 4)            // 314624 < 320K

// ----------------------------------------------------- CDNA5 async copy ----
__device__ __forceinline__ void async_ld_b32(unsigned lds_off, const void* g) {
  // GLOBAL_LOAD_ASYNC_TO_LDS_B32 (GV mode, SADDR=NULL), tracked by ASYNCcnt
  asm volatile("global_load_async_to_lds_b32 %0, %1, off"
               :
               : "v"(lds_off), "v"((unsigned long long)(uintptr_t)g)
               : "memory");
}
__device__ __forceinline__ void wait_async0() {
  asm volatile("s_wait_asynccnt 0" ::: "memory");
}

// 16-element bf16 fragment from two 16B LDS loads
__device__ __forceinline__ v16bf ld2x8(const bf16_t* p0, const bf16_t* p1) {
  v8bf lo = *(const v8bf*)p0;
  v8bf hi = *(const v8bf*)p1;
  return __builtin_shufflevector(lo, hi,
           0,1,2,3,4,5,6,7,8,9,10,11,12,13,14,15);
}

// ---------------------------------------------------------------- kernel ---
__global__ void __launch_bounds__(256, 1)
dys_opt_kernel(const float* __restrict__ c_in, const float* __restrict__ As,
               const float* __restrict__ bs,  const float* __restrict__ Ainv,
               float* __restrict__ out)
{
  extern __shared__ char smem[];
  bf16_t* sA   = (bf16_t*)(smem + OFF_A);     // 128 x 520 bf16
  bf16_t* sAi  = (bf16_t*)(smem + OFF_AI);    // 640 x 136 bf16
  bf16_t* sTbf = (bf16_t*)(smem + OFF_TBF);   // t[0:512] bf16
  bf16_t* sRbf = (bf16_t*)(smem + OFF_RBF);   // r[0:128] bf16
  float*  sSol = (float*) (smem + OFF_SOL);   // sol[640] f32
  float*  sTt  = (float*) (smem + OFF_TTAIL); // t[512:640] f32
  float*  sC   = (float*) (smem + OFF_C);     // c[640] f32
  float*  sB   = (float*) (smem + OFF_BV);    // b[128] f32

  const int tid   = threadIdx.x;
  const int lane  = tid & 31;
  const int wave  = tid >> 5;
  const int half  = lane >> 4;   // K-half select for wave32 fragments
  const int rIL   = lane & 15;   // row-in-lane
  const int batch = blockIdx.x;

  const float* gA  = As   + (size_t)batch * MM * NN;
  const float* gAi = Ainv + (size_t)batch * DDIM * MM;
  const float* gC  = c_in + (size_t)batch * DDIM;
  const float* gB  = bs   + (size_t)batch * MM;

  // ---- async-copy small f32 vectors straight into LDS (ASYNCcnt path) ----
  for (int i = tid; i < DDIM; i += 256)
    async_ld_b32((unsigned)(OFF_C + 4 * i), gC + i);
  if (tid < MM)
    async_ld_b32((unsigned)(OFF_BV + 4 * tid), gB + tid);

  // ---- load + convert matrices fp32 -> bf16 into padded LDS images ----
  // As: 65536 floats = 16384 float4, 64 per thread
  for (int j = 0; j < 64; ++j) {
    int e = (tid + j * 256) * 4;
    int row = e >> 9, col = e & (NN - 1);
    v4f v = *(const v4f*)(gA + e);
    v4bf h;
    h.x = (bf16_t)v.x; h.y = (bf16_t)v.y; h.z = (bf16_t)v.z; h.w = (bf16_t)v.w;
    *(v4bf*)(sA + row * LDA + col) = h;     // 8B ds_store
  }
  // Ainv: 81920 floats = 20480 float4, 80 per thread
  for (int j = 0; j < 80; ++j) {
    int e = (tid + j * 256) * 4;
    int row = e >> 7, col = e & (MM - 1);
    v4f v = *(const v4f*)(gAi + e);
    v4bf h;
    h.x = (bf16_t)v.x; h.y = (bf16_t)v.y; h.z = (bf16_t)v.z; h.w = (bf16_t)v.w;
    *(v4bf*)(sAi + row * LDAI + col) = h;
  }
  // sol0 = 0
  for (int d = tid; d < DDIM; d += 256) sSol[d] = 0.0f;

  wait_async0();
  __syncthreads();

  // =================================================== 50 DYS iterations ===
  for (int it = 0; it < NITER; ++it) {
    float a  = ALPHA_C * powf(1.0f - (float)it * (1.0f / NITER), 1.0f / DECAY_C);
    float c2 = 2.0f - TAU_C * a;   // (2 - tau*a)
    float c1 = 1.0f - TAU_C * a;   // (1 - tau*a)

    // ---- step 1: t = c2*relu(sol) - sol - a*c ; split bf16 head / f32 tail
    for (int d = tid; d < DDIM; d += 256) {
      float s  = sSol[d];
      float p2 = fmaxf(s, 0.0f);
      float t  = c2 * p2 - s - a * sC[d];
      if (d < NN) sTbf[d] = (bf16_t)t;
      else        sTt[d - NN] = t;
    }
    __syncthreads();

    // ---- step 2: r = As @ t[0:512] + t[512:640] - b  (1 row-tile / wave)
    // B fragment: every lane broadcast-loads the SAME vector chunk, so all
    // 16 B-columns are identical; column 0 (lanes 0/16) is the true result.
    // No masking / no exec divergence in the WMMA loop.
    {
      const int m0 = wave * 16;
      v8f acc = {};
      #pragma unroll
      for (int kc = 0; kc < NN / 32; ++kc) {
        const bf16_t* ar = sA + (m0 + rIL) * LDA + kc * 32 + half * 8;
        const bf16_t* xv = sTbf + kc * 32 + half * 8;
        v16bf af = ld2x8(ar, ar + 16);
        v16bf bf = ld2x8(xv, xv + 16);     // LDS broadcast, all lanes
        acc = __builtin_amdgcn_wmma_f32_16x16x32_bf16(
                  false, af, false, bf, (short)0, acc, false, false);
      }
      // column 0 of C lives in lanes 0 (M=g) and 16 (M=8+g), VGPR g
      if (rIL == 0) {
        #pragma unroll
        for (int g = 0; g < 8; ++g) {
          int m = m0 + half * 8 + g;
          float rv = acc[g] + sTt[m] - sB[m];
          sRbf[m] = (bf16_t)rv;
        }
      }
    }
    __syncthreads();

    // ---- step 3: y2 = Ainv @ r ; sol = c1*relu(sol) - a*c - y2
    {
      // hoist the 4 r-chunk B fragments: shared by all 5 row tiles
      v16bf bfr[4];
      #pragma unroll
      for (int kc = 0; kc < 4; ++kc) {
        const bf16_t* xv = sRbf + kc * 32 + half * 8;
        bfr[kc] = ld2x8(xv, xv + 16);      // LDS broadcast, all lanes
      }
      #pragma unroll
      for (int tb = 0; tb < 5; ++tb) {
        const int m0 = (wave + tb * 8) * 16;   // 40 row tiles over 8 waves
        v8f acc = {};
        #pragma unroll
        for (int kc = 0; kc < 4; ++kc) {
          const bf16_t* ar = sAi + (m0 + rIL) * LDAI + kc * 32 + half * 8;
          acc = __builtin_amdgcn_wmma_f32_16x16x32_bf16(
                    false, ld2x8(ar, ar + 16), false, bfr[kc],
                    (short)0, acc, false, false);
        }
        if (rIL == 0) {
          #pragma unroll
          for (int g = 0; g < 8; ++g) {
            int d = m0 + half * 8 + g;
            float s  = sSol[d];
            float p2 = fmaxf(s, 0.0f);
            sSol[d] = c1 * p2 - a * sC[d] - acc[g];
          }
        }
      }
    }
    __syncthreads();
  }

  // ---- output: relu(sol)
  for (int d = tid; d < DDIM; d += 256)
    out[(size_t)batch * DDIM + d] = fmaxf(sSol[d], 0.0f);
}

// ---------------------------------------------------------------- launch ---
extern "C" void kernel_launch(void* const* d_in, const int* in_sizes, int n_in,
                              void* d_out, int out_size, void* d_ws, size_t ws_size,
                              hipStream_t stream) {
  (void)in_sizes; (void)n_in; (void)out_size; (void)d_ws; (void)ws_size;
  const float* c_in = (const float*)d_in[0];   // (256, 640)
  const float* As   = (const float*)d_in[1];   // (256, 128, 512)
  const float* bs   = (const float*)d_in[2];   // (256, 128)
  const float* Ainv = (const float*)d_in[3];   // (256, 640, 128)
  float* out = (float*)d_out;                  // (256, 640)

  hipFuncSetAttribute((const void*)dys_opt_kernel,
                      hipFuncAttributeMaxDynamicSharedMemorySize, LDS_BYTES);
  dys_opt_kernel<<<BATCH, 256, LDS_BYTES, stream>>>(c_in, As, bs, Ainv, out);
}
